// Generator_27479200760074
// MI455X (gfx1250) — matrix operand order repaired and checked by reading, not compiled, not run on hardware
//
#include <hip/hip_runtime.h>
#include <hip/hip_bf16.h>
#include <cmath>

typedef __attribute__((ext_vector_type(16))) _Float16 v16h;
typedef __attribute__((ext_vector_type(8)))  _Float16 v8h;
typedef __attribute__((ext_vector_type(8)))  float    v8f;
typedef __attribute__((ext_vector_type(4)))  float    v4f;

union F16u { v16h v; v8h h[2]; };

#define BB   2
#define SS   2048
#define DD   512
#define HH   8
#define HDIM 64
#define LLAY 8
#define FFD  2048
#define EE   1248
#define BSTOK (BB*SS)   // 4096
#define OUTC 343
#define BHCNT (BB*HH)   // 16
#define CH   64         // attention chunk length
#define NC   (SS/CH)    // 32 chunks

#define BM 64
#define BN 128
#define BK 32
#define LDSA 40   // halves per LDS row (80B) — padded vs 32 to spread banks
#define LDSB 40

// ---------------------------------------------------------------------------
// Generic WMMA GEMM:  C[m, c0+n] = alpha * f(A[m,:]@W[:,n] + bias[n] + addvec[n]) (+ res[m,n])
// Tile: 64(M) x 128(N) x 32(K). Block = 128 threads = 4 waves (wave32).
// A staged row-major [m][k]; B staged TRANSPOSED [n][k]; fragments are
// 2x ds_load_b128 each. LDS double-buffered; one barrier per k-step.
// ---------------------------------------------------------------------------
__global__ __launch_bounds__(128)
void gemm_f16_wmma(const float* __restrict__ A, int lda,
                   const float* __restrict__ W, int ldw,
                   const float* __restrict__ bias,
                   const float* __restrict__ addvec,
                   const float* __restrict__ res, int ldres,
                   float* __restrict__ C, int ldc, int c0,
                   int N, int K, float alpha, int act)
{
    __shared__ __align__(16) _Float16 As[2][BM][LDSA];   // [m][k]
    __shared__ __align__(16) _Float16 Bt[2][BN][LDSB];   // [n][k] (transposed)

    const int tid  = threadIdx.x;
    const int lane = tid & 31;
    const int wave = tid >> 5;        // 0..3
    const int half = lane >> 4;       // 0 or 1
    const int l15  = lane & 15;
    const int bm   = blockIdx.x * BM;
    const int bn   = blockIdx.y * BN;

    const bool fastB = (bn + BN <= N) && ((ldw & 3) == 0);

    auto stageA = [&](int kk, int buf) {
        const int row = tid >> 1;            // 0..63
        const int cb  = (tid & 1) * 16;      // 0 or 16
        const float* src = A + (size_t)(bm + row) * lda + kk + cb;
        const v4f f0 = *(const v4f*)(src);
        const v4f f1 = *(const v4f*)(src + 4);
        const v4f f2 = *(const v4f*)(src + 8);
        const v4f f3 = *(const v4f*)(src + 12);
        v8h h0, h1;
        #pragma unroll
        for (int j = 0; j < 4; ++j) {
            h0[j]     = (_Float16)f0[j];
            h0[j + 4] = (_Float16)f1[j];
            h1[j]     = (_Float16)f2[j];
            h1[j + 4] = (_Float16)f3[j];
        }
        *(v8h*)&As[buf][row][cb]     = h0;
        *(v8h*)&As[buf][row][cb + 8] = h1;
    };

    auto stageB = [&](int kk, int buf) {
        const int kp = tid >> 3;             // 0..15 -> k rows 2kp, 2kp+1
        const int nb = (tid & 7) * 16;       // 16 n values
        const float* r0 = W + (size_t)(kk + 2 * kp) * ldw + bn + nb;
        const float* r1 = r0 + ldw;
        if (fastB) {
            v4f a[4], b[4];
            #pragma unroll
            for (int j = 0; j < 4; ++j) {
                a[j] = *(const v4f*)(r0 + 4 * j);
                b[j] = *(const v4f*)(r1 + 4 * j);
            }
            #pragma unroll
            for (int j = 0; j < 16; ++j) {
                union { _Float16 h[2]; unsigned u; } p;
                p.h[0] = (_Float16)a[j >> 2][j & 3];
                p.h[1] = (_Float16)b[j >> 2][j & 3];
                *(unsigned*)&Bt[buf][nb + j][2 * kp] = p.u;
            }
        } else {
            #pragma unroll
            for (int j = 0; j < 16; ++j) {
                const int n = bn + nb + j;
                union { _Float16 h[2]; unsigned u; } p;
                p.h[0] = (n < N) ? (_Float16)r0[j] : (_Float16)0.0f;
                p.h[1] = (n < N) ? (_Float16)r1[j] : (_Float16)0.0f;
                *(unsigned*)&Bt[buf][nb + j][2 * kp] = p.u;
            }
        }
    };

    v8f acc[8] = {};

    stageA(0, 0);
    stageB(0, 0);
    __syncthreads();

    for (int k0 = 0; k0 < K; k0 += BK) {
        const int cur = (k0 / BK) & 1;
        if (k0 + BK < K) {
            stageA(k0 + BK, cur ^ 1);
            stageB(k0 + BK, cur ^ 1);
        }

        F16u fa;
        fa.h[0] = *(const v8h*)&As[cur][wave * 16 + l15][half * 8];
        fa.h[1] = *(const v8h*)&As[cur][wave * 16 + l15][16 + half * 8];

        F16u fb_cur, fb_nxt;
        fb_cur.h[0] = *(const v8h*)&Bt[cur][l15][half * 8];
        fb_cur.h[1] = *(const v8h*)&Bt[cur][l15][16 + half * 8];
        #pragma unroll
        for (int t = 0; t < 8; ++t) {
            if (t < 7) {
                fb_nxt.h[0] = *(const v8h*)&Bt[cur][(t + 1) * 16 + l15][half * 8];
                fb_nxt.h[1] = *(const v8h*)&Bt[cur][(t + 1) * 16 + l15][16 + half * 8];
            }
            acc[t] = __builtin_amdgcn_wmma_f32_16x16x32_f16(
                         false, fa.v, false, fb_cur.v, (short)0, acc[t], false, false);
            fb_cur.v = fb_nxt.v;
        }
        __syncthreads();
    }

    #pragma unroll
    for (int t = 0; t < 8; ++t) {
        const int n = bn + t * 16 + l15;
        if (n >= N) continue;
        const float bv = bias   ? bias[n]   : 0.0f;
        const float av = addvec ? addvec[n] : 0.0f;
        #pragma unroll
        for (int r = 0; r < 8; ++r) {
            const int m = bm + wave * 16 + half * 8 + r;
            float val = acc[t][r] + bv + av;
            if (act == 1) val = 0.5f * val * (1.0f + erff(val * 0.70710678118654752f));
            val *= alpha;
            if (res) val += res[(size_t)m * ldres + n];
            C[(size_t)m * ldc + c0 + n] = val;
        }
    }
}

// ---------------------------------------------------------------------------
// Multi-field embedding gather -> concat [BS, 1248], scaled by sqrt(d_emb)
// ---------------------------------------------------------------------------
__global__ void embed_concat(const int* __restrict__ inp,
                             const float* __restrict__ t0, const float* __restrict__ t1,
                             const float* __restrict__ t2, const float* __restrict__ t3,
                             const float* __restrict__ t4, const float* __restrict__ t5,
                             const float* __restrict__ t6,
                             float* __restrict__ ec)
{
    const int tok = blockIdx.x;
    const int   offs[8]   = {0,128,384,448,480,992,1120,1248};
    const int   widths[7] = {128,256,64,32,512,128,128};
    const float scl[7]    = {11.3137084990f,16.0f,8.0f,5.6568542495f,
                             22.6274169980f,11.3137084990f,11.3137084990f};
    const float* tabs[7]  = {t0,t1,t2,t3,t4,t5,t6};
    for (int c = threadIdx.x; c < EE; c += blockDim.x) {
        int f = 0;
        while (c >= offs[f + 1]) ++f;
        const int idx = inp[tok * 7 + f];
        ec[(size_t)tok * EE + c] = tabs[f][idx * widths[f] + (c - offs[f])] * scl[f];
    }
}

// positional-encoding vector (reference indexes pe at position S)
__global__ void compute_pe(float* __restrict__ pe)
{
    const int d = blockIdx.x * blockDim.x + threadIdx.x;
    if (d >= DD) return;
    const int i = d >> 1;
    const float div = expf((float)(2 * i) * (-logf(10000.0f) / (float)DD));
    const float ang = (float)SS * div;
    pe[d] = (d & 1) ? cosf(ang) : sinf(ang);
}

// RoPE + phi (elu+1) on q,k in place. One thread per (token, head, pair i<32).
__global__ void rope_phi(float* __restrict__ q, float* __restrict__ k)
{
    const int gid = blockIdx.x * blockDim.x + threadIdx.x;
    if (gid >= BSTOK * HH * 32) return;
    const int i   = gid & 31;
    const int h   = (gid >> 5) & 7;
    const int tok = gid >> 8;
    const int s   = tok & (SS - 1);
    const float freq = expf((float)(2 * i) * (-logf(10000.0f) / (float)HDIM));
    const float ang  = (float)s * freq;
    const float c = cosf(ang), sn = sinf(ang);
    const size_t base = (size_t)tok * DD + h * HDIM;
    const float q1 = q[base + i], q2 = q[base + i + 32];
    const float k1 = k[base + i], k2 = k[base + i + 32];
    const float qr1 = q1 * c - q2 * sn, qr2 = q2 * c + q1 * sn;
    const float kr1 = k1 * c - k2 * sn, kr2 = k2 * c + k1 * sn;
    q[base + i]      = qr1 > 0.f ? qr1 + 1.f : expf(qr1);
    q[base + i + 32] = qr2 > 0.f ? qr2 + 1.f : expf(qr2);
    k[base + i]      = kr1 > 0.f ? kr1 + 1.f : expf(kr1);
    k[base + i + 32] = kr2 > 0.f ? kr2 + 1.f : expf(kr2);
}

// ---------------------------------------------------------------------------
// Chunked causal linear attention (3 phases).
// Phase 1: per (b,h,chunk): local KV = K_chunk^T @ V_chunk (64x64), local ksum.
// ---------------------------------------------------------------------------
__global__ __launch_bounds__(256)
void attn_chunk_kv(const float* __restrict__ k, const float* __restrict__ v,
                   float* __restrict__ skv, float* __restrict__ ksum)
{
    const int bh = blockIdx.x, c = blockIdx.y;
    const int b = bh >> 3, h = bh & 7;
    const int tid = threadIdx.x;
    __shared__ float Ks[CH][HDIM + 1], Vs[CH][HDIM + 1];
    for (int idx = tid; idx < CH * HDIM; idx += 256) {
        const int u = idx >> 6, d = idx & 63;
        const size_t base = ((size_t)(b * SS + c * CH + u)) * DD + h * HDIM;
        Ks[u][d] = k[base + d];
        Vs[u][d] = v[base + d];
    }
    __syncthreads();
    const int d  = tid >> 2;           // 0..63
    const int mb = (tid & 3) * 16;     // 16 m values
    float acc[16] = {};
    for (int u = 0; u < CH; ++u) {
        const float kd = Ks[u][d];
        #pragma unroll
        for (int j = 0; j < 16; ++j) acc[j] += kd * Vs[u][mb + j];
    }
    const size_t ob = (((size_t)bh * NC) + c) * (HDIM * HDIM) + d * HDIM + mb;
    #pragma unroll
    for (int j = 0; j < 16; ++j) skv[ob + j] = acc[j];
    if (tid < HDIM) {
        float s = 0.f;
        for (int u = 0; u < CH; ++u) s += Ks[u][tid];
        ksum[((size_t)bh * NC + c) * HDIM + tid] = s;
    }
}

// Phase 2: exclusive prefix over chunks (32 sequential iterations per (b,h)).
__global__ __launch_bounds__(256)
void attn_prefix(const float* __restrict__ skv, const float* __restrict__ ksum,
                 float* __restrict__ pkv, float* __restrict__ pksum)
{
    const int bh  = blockIdx.x;
    const int tid = threadIdx.x;
    float run[16];
    #pragma unroll
    for (int j = 0; j < 16; ++j) run[j] = 0.f;
    float krun = 0.f;
    for (int c = 0; c < NC; ++c) {
        const size_t base = ((size_t)bh * NC + c) * (HDIM * HDIM) + tid * 16;
        if (c + 1 < NC)  // cover next chunk's lines while we chew this one
            __builtin_prefetch(&skv[base + HDIM * HDIM], 0, 0);
        #pragma unroll
        for (int j = 0; j < 16; ++j) {
            pkv[base + j] = run[j];
            run[j] += skv[base + j];
        }
        if (tid < HDIM) {
            const size_t kb = ((size_t)bh * NC + c) * HDIM + tid;
            pksum[kb] = krun;
            krun += ksum[kb];
        }
    }
}

// ---------------------------------------------------------------------------
// Phase 3 (WMMA): per (b,h,chunk), with 64x64 f16 tiles in LDS:
//   P = Q @ K^T            (Kc row-major == transposed-B layout for K^T)
//   P = tril(P)            (masked while spilling accumulators to LDS)
//   O = P @ V + Q @ S_pref (two accumulated WMMA products)
//   z_t = q_t.ksum_pref + rowsum(P_t);  out = O / (z + eps)
// Block = 128 threads = 4 waves; wave w owns output rows 16w..16w+15.
// ---------------------------------------------------------------------------
__global__ __launch_bounds__(128)
void attn_chunk_out(const float* __restrict__ q, const float* __restrict__ k,
                    const float* __restrict__ v, const float* __restrict__ pkv,
                    const float* __restrict__ pksum, float* __restrict__ o)
{
    const int bh = blockIdx.x, c = blockIdx.y;
    const int b = bh >> 3, h = bh & 7;
    const int tid = threadIdx.x;
    const int lane = tid & 31;
    const int wave = tid >> 5;
    const int half = lane >> 4;
    const int l15  = lane & 15;

    __shared__ __align__(16) _Float16 Qc[CH][72];     // [t][d]
    __shared__ __align__(16) _Float16 Kc[CH][72];     // [u][d] (Bt for Q@K^T)
    __shared__ __align__(16) _Float16 Vt[HDIM][72];   // [m][u] (transposed V)
    __shared__ __align__(16) _Float16 PKt[HDIM][72];  // [m][d] (transposed prefix KV)
    __shared__ __align__(16) _Float16 Ps[CH][72];     // [t][u] masked scores
    __shared__ float zrow[CH];
    __shared__ float pks[HDIM];

    // ---- stage Q,K row-major (thread -> row tid>>1, 32 consecutive d)
    {
        const int r  = tid >> 1;
        const int db = (tid & 1) * 32;
        const size_t gb = ((size_t)(b * SS + c * CH + r)) * DD + h * HDIM + db;
        #pragma unroll
        for (int p = 0; p < 4; ++p) {
            const v4f a0 = *(const v4f*)(q + gb + 8 * p);
            const v4f a1 = *(const v4f*)(q + gb + 8 * p + 4);
            const v4f b0 = *(const v4f*)(k + gb + 8 * p);
            const v4f b1 = *(const v4f*)(k + gb + 8 * p + 4);
            v8h hq, hk;
            #pragma unroll
            for (int j = 0; j < 4; ++j) {
                hq[j] = (_Float16)a0[j]; hq[j + 4] = (_Float16)a1[j];
                hk[j] = (_Float16)b0[j]; hk[j + 4] = (_Float16)b1[j];
            }
            *(v8h*)&Qc[r][db + 8 * p] = hq;
            *(v8h*)&Kc[r][db + 8 * p] = hk;
        }
    }
    // ---- stage V, prefix-KV transposed (thread -> row-pair tid>>2, 16 cols)
    {
        const int up = tid >> 2;            // 0..31 -> rows 2up, 2up+1
        const int mb = (tid & 3) * 16;
        const float* r0 = v + ((size_t)(b * SS + c * CH + 2 * up)) * DD + h * HDIM + mb;
        const float* r1 = r0 + DD;
        const float* s0 = pkv + (((size_t)bh * NC) + c) * (HDIM * HDIM) + (2 * up) * HDIM + mb;
        const float* s1 = s0 + HDIM;
        v4f va[4], vb[4], pa[4], pbv[4];
        #pragma unroll
        for (int p = 0; p < 4; ++p) {
            va[p] = *(const v4f*)(r0 + 4 * p);
            vb[p] = *(const v4f*)(r1 + 4 * p);
            pa[p] = *(const v4f*)(s0 + 4 * p);
            pbv[p]= *(const v4f*)(s1 + 4 * p);
        }
        #pragma unroll
        for (int j = 0; j < 16; ++j) {
            union { _Float16 h[2]; unsigned u; } p1, p2;
            p1.h[0] = (_Float16)va[j >> 2][j & 3];
            p1.h[1] = (_Float16)vb[j >> 2][j & 3];
            p2.h[0] = (_Float16)pa[j >> 2][j & 3];
            p2.h[1] = (_Float16)pbv[j >> 2][j & 3];
            *(unsigned*)&Vt[mb + j][2 * up]  = p1.u;
            *(unsigned*)&PKt[mb + j][2 * up] = p2.u;
        }
    }
    if (tid < HDIM) pks[tid] = pksum[((size_t)bh * NC + c) * HDIM + tid];
    __syncthreads();

    // ---- P = Q @ K^T  (wave w: rows 16w..16w+15; 4 u-tiles; K=64 -> 2 steps)
    v8f pacc[4] = {};
    #pragma unroll
    for (int ks = 0; ks < 2; ++ks) {
        F16u fa;
        fa.h[0] = *(const v8h*)&Qc[wave * 16 + l15][ks * 32 + half * 8];
        fa.h[1] = *(const v8h*)&Qc[wave * 16 + l15][ks * 32 + 16 + half * 8];
        #pragma unroll
        for (int tt = 0; tt < 4; ++tt) {
            F16u fb;
            fb.h[0] = *(const v8h*)&Kc[tt * 16 + l15][ks * 32 + half * 8];
            fb.h[1] = *(const v8h*)&Kc[tt * 16 + l15][ks * 32 + 16 + half * 8];
            pacc[tt] = __builtin_amdgcn_wmma_f32_16x16x32_f16(
                           false, fa.v, false, fb.v, (short)0, pacc[tt], false, false);
        }
    }
    // ---- causal mask while spilling P to LDS (f16)
    #pragma unroll
    for (int tt = 0; tt < 4; ++tt) {
        const int u = tt * 16 + l15;
        #pragma unroll
        for (int r = 0; r < 8; ++r) {
            const int t = wave * 16 + half * 8 + r;
            Ps[t][u] = (u <= t) ? (_Float16)pacc[tt][r] : (_Float16)0.0f;
        }
    }
    __syncthreads();

    // ---- z_t = rowsum(P_t) + q_t . ksum_prefix
    if (tid < CH) {
        float zz = 0.f;
        for (int u = 0; u < CH; ++u) zz += (float)Ps[tid][u];
        for (int dd = 0; dd < HDIM; ++dd) zz += (float)Qc[tid][dd] * pks[dd];
        zrow[tid] = zz;
    }

    // ---- O = P @ V + Q @ S_prefix (accumulated)
    v8f oacc[4] = {};
    #pragma unroll
    for (int ks = 0; ks < 2; ++ks) {
        F16u fp, fq;
        fp.h[0] = *(const v8h*)&Ps[wave * 16 + l15][ks * 32 + half * 8];
        fp.h[1] = *(const v8h*)&Ps[wave * 16 + l15][ks * 32 + 16 + half * 8];
        fq.h[0] = *(const v8h*)&Qc[wave * 16 + l15][ks * 32 + half * 8];
        fq.h[1] = *(const v8h*)&Qc[wave * 16 + l15][ks * 32 + 16 + half * 8];
        #pragma unroll
        for (int tt = 0; tt < 4; ++tt) {
            F16u fv, fpk;
            fv.h[0]  = *(const v8h*)&Vt[tt * 16 + l15][ks * 32 + half * 8];
            fv.h[1]  = *(const v8h*)&Vt[tt * 16 + l15][ks * 32 + 16 + half * 8];
            fpk.h[0] = *(const v8h*)&PKt[tt * 16 + l15][ks * 32 + half * 8];
            fpk.h[1] = *(const v8h*)&PKt[tt * 16 + l15][ks * 32 + 16 + half * 8];
            oacc[tt] = __builtin_amdgcn_wmma_f32_16x16x32_f16(
                           false, fp.v, false, fv.v, (short)0, oacc[tt], false, false);
            oacc[tt] = __builtin_amdgcn_wmma_f32_16x16x32_f16(
                           false, fq.v, false, fpk.v, (short)0, oacc[tt], false, false);
        }
    }
    __syncthreads();   // zrow visible to all

    #pragma unroll
    for (int tt = 0; tt < 4; ++tt) {
        const int m = tt * 16 + l15;
        #pragma unroll
        for (int r = 0; r < 8; ++r) {
            const int t = wave * 16 + half * 8 + r;
            const size_t gb = ((size_t)(b * SS + c * CH + t)) * DD + h * HDIM + m;
            o[gb] = oacc[tt][r] / (zrow[t] + 1e-6f);
        }
    }
}

// LayerNorm over D=512, one block (128 threads) per row; in-place safe.
__global__ __launch_bounds__(128)
void layernorm(const float* __restrict__ x, const float* __restrict__ g,
               const float* __restrict__ b, float* __restrict__ o)
{
    const int row = blockIdx.x;
    __shared__ float red[128];
    const float* xr = x + (size_t)row * DD;
    float v[4];
    float s = 0.f;
    #pragma unroll
    for (int j = 0; j < 4; ++j) { v[j] = xr[threadIdx.x + j * 128]; s += v[j]; }
    red[threadIdx.x] = s; __syncthreads();
    for (int st = 64; st > 0; st >>= 1) {
        if (threadIdx.x < st) red[threadIdx.x] += red[threadIdx.x + st];
        __syncthreads();
    }
    const float mean = red[0] / (float)DD;
    __syncthreads();
    s = 0.f;
    #pragma unroll
    for (int j = 0; j < 4; ++j) { const float d = v[j] - mean; s += d * d; }
    red[threadIdx.x] = s; __syncthreads();
    for (int st = 64; st > 0; st >>= 1) {
        if (threadIdx.x < st) red[threadIdx.x] += red[threadIdx.x + st];
        __syncthreads();
    }
    const float inv = rsqrtf(red[0] / (float)DD + 1e-5f);
    #pragma unroll
    for (int j = 0; j < 4; ++j) {
        const int c = threadIdx.x + j * 128;
        o[(size_t)row * DD + c] = (v[j] - mean) * inv * g[c] + b[c];
    }
}

// concat [x (512) | emb_type[target] * sqrt(32) (32)] -> [BS, 544]
__global__ void concat_xc(const float* __restrict__ x, const int* __restrict__ target,
                          const float* __restrict__ embt, float* __restrict__ xc)
{
    const int tok = blockIdx.x;
    for (int c = threadIdx.x; c < 544; c += blockDim.x) {
        float val;
        if (c < 512) val = x[(size_t)tok * DD + c];
        else {
            const int idx = target[tok * 7 + 3];
            val = embt[idx * 32 + (c - 512)] * 5.6568542495f;
        }
        xc[(size_t)tok * 544 + c] = val;
    }
}

// ---------------------------------------------------------------------------
extern "C" void kernel_launch(void* const* d_in, const int* in_sizes, int n_in,
                              void* d_out, int out_size, void* d_ws, size_t ws_size,
                              hipStream_t stream)
{
    (void)in_sizes; (void)n_in; (void)out_size; (void)ws_size;

    const int*   inputs = (const int*)d_in[0];
    const int*   target = (const int*)d_in[1];
    const float* emb[7];
    for (int i = 0; i < 7; ++i) emb[i] = (const float*)d_in[2 + i];
    const float* in_W  = (const float*)d_in[9];
    const float* in_b  = (const float*)d_in[10];
    const float* Wq    = (const float*)d_in[11];
    const float* Wk    = (const float*)d_in[12];
    const float* Wv    = (const float*)d_in[13];
    const float* Wo    = (const float*)d_in[14];
    const float* W1    = (const float*)d_in[15];
    const float* W2    = (const float*)d_in[16];
    const float* bq    = (const float*)d_in[17];
    const float* bk    = (const float*)d_in[18];
    const float* bv    = (const float*)d_in[19];
    const float* bo    = (const float*)d_in[20];
    const float* b1    = (const float*)d_in[21];
    const float* b2    = (const float*)d_in[22];
    const float* ln1_b = (const float*)d_in[23];
    const float* ln2_b = (const float*)d_in[24];
    const float* norm_b= (const float*)d_in[25];
    const float* catb  = (const float*)d_in[26];
    const float* ln1_g = (const float*)d_in[27];
    const float* ln2_g = (const float*)d_in[28];
    const float* norm_g= (const float*)d_in[29];
    const float* pW[7];
    const float* pb[7];
    for (int i = 0; i < 7; ++i) { pW[i] = (const float*)d_in[30 + 2 * i];
                                  pb[i] = (const float*)d_in[31 + 2 * i]; }
    const float* catW = (const float*)d_in[44];

    float* out = (float*)d_out;

    // workspace layout (floats)
    float* ws  = (float*)d_ws;
    float* EC  = ws;                               // [BS,1248] (reused: attn prefix, XC, XC2)
    float* X   = EC  + (size_t)BSTOK * EE;         // [BS,512]
    float* Q   = X   + (size_t)BSTOK * DD;
    float* Kb  = Q   + (size_t)BSTOK * DD;
    float* Vb  = Kb  + (size_t)BSTOK * DD;
    float* ATT = Vb  + (size_t)BSTOK * DD;
    float* HID = ATT + (size_t)BSTOK * DD;         // [BS,2048] (reused: attn chunk KV)
    float* PE  = HID + (size_t)BSTOK * FFD;        // [512]
    // attention scratch (aliases; no temporal overlap with owners)
    float* SKV = HID;                              // [16][NC][64*64] = 2.10M floats (HID has 8.39M)
    float* KS  = SKV + (size_t)BHCNT * NC * HDIM * HDIM;   // [16][NC][64]
    float* PKV = EC;                               // [16][NC][64*64]  (EC has 5.11M)
    float* PKS = PKV + (size_t)BHCNT * NC * HDIM * HDIM;   // [16][NC][64]
    // head scratch (EC dead again by then)
    float* XC  = EC;                               // [BS,544]
    float* XC2 = EC + (size_t)BSTOK * 544;         // [BS,512]

    const dim3 blk(128);
    const dim3 gD (BSTOK / BM, DD  / BN);          // N=512  -> y=4
    const dim3 gFF(BSTOK / BM, FFD / BN);          // N=2048 -> y=16
    const dim3 gAT(BHCNT, NC);

    compute_pe<<<1, 512, 0, stream>>>(PE);
    embed_concat<<<BSTOK, 128, 0, stream>>>(inputs, emb[0], emb[1], emb[2], emb[3],
                                            emb[4], emb[5], emb[6], EC);
    // x = concat_embs @ in_W + in_b + pe
    gemm_f16_wmma<<<gD, blk, 0, stream>>>(EC, EE, in_W, DD, in_b, PE,
                                          nullptr, 0, X, DD, 0, DD, EE, 1.0f, 0);

    for (int l = 0; l < LLAY; ++l) {
        const float* Wql = Wq + (size_t)l * DD * DD;
        const float* Wkl = Wk + (size_t)l * DD * DD;
        const float* Wvl = Wv + (size_t)l * DD * DD;
        const float* Wol = Wo + (size_t)l * DD * DD;
        const float* W1l = W1 + (size_t)l * DD * FFD;
        const float* W2l = W2 + (size_t)l * FFD * DD;

        gemm_f16_wmma<<<gD, blk, 0, stream>>>(X, DD, Wql, DD, bq + l * DD, nullptr,
                                              nullptr, 0, Q,  DD, 0, DD, DD, 1.0f, 0);
        gemm_f16_wmma<<<gD, blk, 0, stream>>>(X, DD, Wkl, DD, bk + l * DD, nullptr,
                                              nullptr, 0, Kb, DD, 0, DD, DD, 1.0f, 0);
        gemm_f16_wmma<<<gD, blk, 0, stream>>>(X, DD, Wvl, DD, bv + l * DD, nullptr,
                                              nullptr, 0, Vb, DD, 0, DD, DD, 1.0f, 0);

        rope_phi<<<(BSTOK * HH * 32 + 255) / 256, 256, 0, stream>>>(Q, Kb);

        // chunked causal linear attention
        attn_chunk_kv<<<gAT, 256, 0, stream>>>(Kb, Vb, SKV, KS);
        attn_prefix  <<<BHCNT, 256, 0, stream>>>(SKV, KS, PKV, PKS);
        attn_chunk_out<<<gAT, blk, 0, stream>>>(Q, Kb, Vb, PKV, PKS, ATT);

        // x = x + attn @ Wo + bo
        gemm_f16_wmma<<<gD, blk, 0, stream>>>(ATT, DD, Wol, DD, bo + l * DD, nullptr,
                                              X, DD, X, DD, 0, DD, DD, 1.0f, 0);
        layernorm<<<BSTOK, 128, 0, stream>>>(X, ln1_g + l * DD, ln1_b + l * DD, X);

        // hid = gelu(x @ W1 + b1) ; x = x + hid @ W2 + b2
        gemm_f16_wmma<<<gFF, blk, 0, stream>>>(X, DD, W1l, FFD, b1 + l * FFD, nullptr,
                                               nullptr, 0, HID, FFD, 0, FFD, DD, 1.0f, 1);
        gemm_f16_wmma<<<gD, blk, 0, stream>>>(HID, FFD, W2l, DD, b2 + l * DD, nullptr,
                                              X, DD, X, DD, 0, DD, FFD, 1.0f, 0);
        layernorm<<<BSTOK, 128, 0, stream>>>(X, ln2_g + l * DD, ln2_b + l * DD, X);
    }

    layernorm<<<BSTOK, 128, 0, stream>>>(X, norm_g, norm_b, X);

    // head: type projection from pre-concat x
    static const int voc[7]  = {56, 135, 18, 4, 87, 18, 25};
    static const int coff[7] = {0, 56, 191, 209, 213, 300, 318};
    {
        const dim3 g(BSTOK / BM, (voc[3] + BN - 1) / BN);
        gemm_f16_wmma<<<g, blk, 0, stream>>>(X, DD, pW[3], voc[3], pb[3], nullptr,
                                             nullptr, 0, out, OUTC, coff[3],
                                             voc[3], DD, sqrtf((float)voc[3]), 0);
    }
    // xc = [x | emb_type(target)*sqrt(32)] @ catW + catb
    concat_xc<<<BSTOK, 128, 0, stream>>>(X, target, emb[3], XC);
    gemm_f16_wmma<<<gD, blk, 0, stream>>>(XC, 544, catW, DD, catb, nullptr,
                                          nullptr, 0, XC2, DD, 0, DD, 544, 1.0f, 0);
    // remaining 6 projections from xc
    for (int i = 0; i < 7; ++i) {
        if (i == 3) continue;
        const dim3 g(BSTOK / BM, (voc[i] + BN - 1) / BN);
        gemm_f16_wmma<<<g, blk, 0, stream>>>(XC2, DD, pW[i], voc[i], pb[i], nullptr,
                                             nullptr, 0, out, OUTC, coff[i],
                                             voc[i], DD, sqrtf((float)voc[i]), 0);
    }
}
